// MultiHeadAttention_33122787786782
// MI455X (gfx1250) — compile-verified, matrix operand order checked by reading, not run on hardware
//
#include <hip/hip_runtime.h>
#include <hip/hip_bf16.h>

// ---------------------------------------------------------------------------
// Relative-position multi-head attention (Music-Transformer skewing) for
// MI455X / gfx1250.  All GEMMs run through V_WMMA_F32_16X16X32_BF16 with
// fp32 accumulation.  bf16->bf16 LDS staging uses the CDNA5 async path
// (GLOBAL_LOAD_ASYNC_TO_LDS_B128, ASYNCcnt) with double-buffered chunks so
// the async engine prefetches behind the WMMAs.  Shapes: B=4, L=384,
// DM=1024, H=16, D=64, M=63.
// ---------------------------------------------------------------------------

typedef __bf16 bf16;
typedef __attribute__((ext_vector_type(16))) __bf16 v16bf;
typedef __attribute__((ext_vector_type(8)))  float  v8f;

#define B_   4
#define L_   384
#define DM_  1024
#define H_   16
#define D_   64
#define MREL 63

__device__ __forceinline__ int frag_k(int e, int lane) {
  // 16-bit A/B matrix VGPR layout (ISA 7.12.2): element e of the v16bf holds
  // k = (e>=8?16:0) + (lane>=16?8:0) + (e&7)
  return ((e & 8) << 1) | ((lane & 16) >> 1) | (e & 7);
}

__device__ __forceinline__ v16bf load_frag(const bf16* base, int ld, int lane) {
  // A[m][k] (or B staged transposed as Bt[n][k]): lane holds row m = lane&15.
  const bf16* row = base + (lane & 15) * ld;
  v16bf f;
#pragma unroll
  for (int e = 0; e < 16; ++e) f[e] = row[frag_k(e, lane)];
  return f;
}

__device__ __forceinline__ v16bf load_frag_f32(const float* base, int ld, int lane) {
  const float* row = base + (lane & 15) * ld;
  v16bf f;
#pragma unroll
  for (int e = 0; e < 16; ++e) f[e] = (bf16)row[frag_k(e, lane)];
  return f;
}

__device__ __forceinline__ v8f wmma_bf16(v16bf a, v16bf b, v8f c) {
  return __builtin_amdgcn_wmma_f32_16x16x32_bf16(false, a, false, b,
                                                 (short)0, c, false, false);
}

__device__ __forceinline__ v8f zero8() {
  v8f z;
#pragma unroll
  for (int i = 0; i < 8; ++i) z[i] = 0.f;
  return z;
}

// ---- CDNA5 async global->LDS copy (16 bytes / lane, tracked by ASYNCcnt) --
// lds_byte_off is relative to the workgroup's (all-dynamic) LDS allocation.
__device__ __forceinline__ void async_b128(unsigned lds_byte_off,
                                           const void* gaddr) {
  asm volatile("global_load_async_to_lds_b128 %0, %1, off"
               :: "v"(lds_byte_off),
                  "v"((unsigned long long)gaddr)
               : "memory");
}
__device__ __forceinline__ void wait_async0() {
  asm volatile("s_wait_asynccnt 0x0" ::: "memory");
}
// Leave the newest 2 async ops (next chunk) in flight; oldest 2 are done.
__device__ __forceinline__ void wait_async2() {
  asm volatile("s_wait_asynccnt 0x2" ::: "memory");
}

// ---------------------------------------------------------------------------
// Kernel 1: fused QKV projection.  X(1536x1024) @ W(1024x1024), bf16 WMMA.
// grid (DM/128, 1536/128, 3); block 256 (8 waves).  z selects q/k/v.
// Both tiles need fp32->bf16 conversion, so staging stays on the VALU path;
// next-tile global_prefetch_b8 hides HBM latency behind the WMMAs.
// Q,K stored (B,H,L,D); V stored transposed (B,H,D,L).
// ---------------------------------------------------------------------------
__global__ __launch_bounds__(256) void proj_kernel(
    const float* __restrict__ q, const float* __restrict__ k,
    const float* __restrict__ v,
    const float* __restrict__ Wq, const float* __restrict__ Wk,
    const float* __restrict__ Wv,
    bf16* __restrict__ Qh, bf16* __restrict__ Kh, bf16* __restrict__ Vt) {
  __shared__ bf16 As[128 * 40];   // A[m][k], k-slice of 32, stride 40
  __shared__ bf16 Bt[128 * 40];   // B staged transposed: Bt[n][k]

  const int z = blockIdx.z;
  const float* X = (z == 0) ? q : (z == 1) ? k : v;
  const float* W = (z == 0) ? Wq : (z == 1) ? Wk : Wv;

  const int row0 = blockIdx.y * 128;
  const int col0 = blockIdx.x * 128;
  const int tid  = threadIdx.x;
  const int lane = tid & 31;
  const int w    = tid >> 5;
  const int wr   = w >> 2;   // 64-row block within tile
  const int wc   = w & 3;    // 32-col block within tile

  v8f acc[4][2];
#pragma unroll
  for (int i = 0; i < 4; ++i)
#pragma unroll
    for (int j = 0; j < 2; ++j) acc[i][j] = zero8();

  for (int k0 = 0; k0 < DM_; k0 += 32) {
    {  // A tile 128x32, fp32 -> bf16
      int r = tid >> 1, s = (tid & 1) * 16;
      const float* src = X + (size_t)(row0 + r) * DM_ + k0 + s;
#pragma unroll
      for (int i = 0; i < 16; ++i) As[r * 40 + s + i] = (bf16)src[i];
      if (k0 + 32 < DM_) __builtin_prefetch(src + 32, 0, 0);
    }
    {  // B tile 32x128, transpose into Bt[n][k]
      int kk = tid >> 3, s = (tid & 7) * 16;
      const float* src = W + (size_t)(k0 + kk) * DM_ + col0 + s;
#pragma unroll
      for (int i = 0; i < 16; ++i) Bt[(s + i) * 40 + kk] = (bf16)src[i];
      if (k0 + 32 < DM_) __builtin_prefetch(src + 32 * DM_, 0, 0);
    }
    __syncthreads();
#pragma unroll
    for (int i = 0; i < 4; ++i) {
      v16bf fa = load_frag(As + (wr * 64 + i * 16) * 40, 40, lane);
#pragma unroll
      for (int j = 0; j < 2; ++j) {
        v16bf fb = load_frag(Bt + (wc * 32 + j * 16) * 40, 40, lane);
        acc[i][j] = wmma_bf16(fa, fb, acc[i][j]);
      }
    }
    __syncthreads();
  }

  const int hh = (lane & 16) >> 1;  // +8 rows for upper half-wave
  const int nl = lane & 15;
#pragma unroll
  for (int i = 0; i < 4; ++i)
#pragma unroll
    for (int j = 0; j < 2; ++j)
#pragma unroll
      for (int r = 0; r < 8; ++r) {
        int Rg = row0 + wr * 64 + i * 16 + r + hh;
        int Cg = col0 + wc * 32 + j * 16 + nl;
        int bb = Rg / L_, ll = Rg % L_;
        int h = Cg >> 6, d = Cg & 63;
        bf16 val = (bf16)acc[i][j][r];
        if (z == 0)      Qh[(((size_t)(bb * H_ + h)) * L_ + ll) * D_ + d] = val;
        else if (z == 1) Kh[(((size_t)(bb * H_ + h)) * L_ + ll) * D_ + d] = val;
        else             Vt[(((size_t)(bb * H_ + h)) * D_ + d) * L_ + ll] = val;
      }
}

// ---------------------------------------------------------------------------
// Kernel 2: attention core.  One block = (b, h, 64 query rows).
// LDS byte-offset map (all-dynamic shared, base offset 0):
//   Qs   [64x72 bf16] @ 0        Ks0  [64x72 bf16] @ 9216
//   Ks1  [64x72 bf16] @ 18432    Vts0 [64x72 bf16] @ 27648
//   Vts1 [64x72 bf16] @ 36864    KRs  [64x72 bf16] @ 46080
//   VRTs [64x72 bf16] @ 55296    attn [64x392 bf16] @ 64512
//   rel  [64x64 f32]  @ 114688   padm [384 f32]     @ 131072
//   red  [128 f32]    @ 132608                       (total 133120 B)
// K / Vt chunks are double-buffered via GLOBAL_LOAD_ASYNC_TO_LDS_B128:
// chunk c+1 is in flight (ASYNCcnt=2) while chunk c feeds the WMMAs.
// ---------------------------------------------------------------------------
#define QS_OFF    0u
#define KS0_OFF   9216u
#define KS1_OFF   18432u
#define VTS0_OFF  27648u
#define VTS1_OFF  36864u
#define KRS_OFF   46080u
#define VRTS_OFF  55296u
#define ATTN_OFF  64512u
#define REL_OFF   114688u
#define PADM_OFF  131072u
#define RED_OFF   132608u
#define SMEM2_BYTES 133120u

__global__ __launch_bounds__(256) void attn_kernel(
    const bf16* __restrict__ Qh, const bf16* __restrict__ Kh,
    const bf16* __restrict__ Vt,
    const float* __restrict__ key_rel, const float* __restrict__ value_rel,
    const float* __restrict__ pad_mask,
    bf16* __restrict__ O, float* __restrict__ attn_out) {
  extern __shared__ unsigned char smem[];
  bf16*  Qs     = (bf16*)(smem + QS_OFF);
  bf16*  Ks0    = (bf16*)(smem + KS0_OFF);
  bf16*  Ks1    = (bf16*)(smem + KS1_OFF);
  bf16*  Vts0   = (bf16*)(smem + VTS0_OFF);
  bf16*  Vts1   = (bf16*)(smem + VTS1_OFF);
  bf16*  KRs    = (bf16*)(smem + KRS_OFF);
  bf16*  VRTs   = (bf16*)(smem + VRTS_OFF);
  bf16*  attn_s = (bf16*)(smem + ATTN_OFF);
  float* rel_s  = (float*)(smem + REL_OFF);
  float* padm   = (float*)(smem + PADM_OFF);
  float* red    = (float*)(smem + RED_OFF);

  const int lt = blockIdx.x;            // query-row tile (64 rows)
  const int h  = blockIdx.y;
  const int b  = blockIdx.z;
  const int bh = b * H_ + h;
  const int tid  = threadIdx.x;
  const int lane = tid & 31;
  const int w    = tid >> 5;
  const int wrt  = w >> 1;              // row tile 0..3 within the 64 rows
  const int wch  = w & 1;               // column half
  const int hh   = (lane & 16) >> 1;
  const int nl   = lane & 15;

  const int sr = tid >> 2;              // staging row 0..63
  const int ss = (tid & 3) * 16;        // staging col 0..48 (16 bf16 = 32B)
  const unsigned soff = (unsigned)(sr * 72 + ss) * 2;  // LDS byte offset

  // ---- stage Q (async) / key_rel / value_relT / pad_mask -----------------
  {
    const bf16* src = Qh + ((size_t)bh * L_ + lt * 64 + sr) * D_ + ss;
    async_b128(QS_OFF + soff, src);
    async_b128(QS_OFF + soff + 16u, src + 8);
    if (sr < MREL) {
      const float* kr = key_rel + ((size_t)h * MREL + sr) * D_ + ss;
      const float* vr = value_rel + ((size_t)h * MREL + sr) * D_ + ss;
#pragma unroll
      for (int i = 0; i < 16; ++i) {
        KRs[sr * 72 + ss + i]    = (bf16)kr[i];
        VRTs[(ss + i) * 72 + sr] = (bf16)vr[i];
      }
    } else {
#pragma unroll
      for (int i = 0; i < 16; ++i) {
        KRs[sr * 72 + ss + i]    = (bf16)0.f;
        VRTs[(ss + i) * 72 + sr] = (bf16)0.f;
      }
    }
  }
  for (int j = tid; j < L_; j += 256) padm[j] = pad_mask[(size_t)b * L_ + j];
  wait_async0();
  __syncthreads();

  // ---- rel[i][m] = Q . key_rel^T  (half the waves) -----------------------
  if (wch == 0) {
    v8f rc[4];
#pragma unroll
    for (int n = 0; n < 4; ++n) rc[n] = zero8();
#pragma unroll
    for (int n = 0; n < 4; ++n)
#pragma unroll
      for (int ks = 0; ks < 2; ++ks) {
        v16bf fa = load_frag(Qs + (wrt * 16) * 72 + ks * 32, 72, lane);
        v16bf fb = load_frag(KRs + (n * 16) * 72 + ks * 32, 72, lane);
        rc[n] = wmma_bf16(fa, fb, rc[n]);
      }
#pragma unroll
    for (int n = 0; n < 4; ++n)
#pragma unroll
      for (int r = 0; r < 8; ++r)
        rel_s[(wrt * 16 + r + hh) * 64 + n * 16 + nl] = rc[n][r];
  }
  __syncthreads();

  // ---- scores = Q K^T over 6 chunks of 64 keys ---------------------------
  // Double-buffered async staging: chunk c+1 streams while chunk c computes.
  v8f sc[12];
#pragma unroll
  for (int i = 0; i < 12; ++i) sc[i] = zero8();

  {  // prologue: chunk 0 -> Ks0
    const bf16* src = Kh + ((size_t)bh * L_ + 0 * 64 + sr) * D_ + ss;
    async_b128(KS0_OFF + soff, src);
    async_b128(KS0_OFF + soff + 16u, src + 8);
  }
  for (int c = 0; c < 6; ++c) {
    if (c + 1 < 6) {  // stage chunk c+1 into the alternate buffer
      unsigned dst = (c & 1) ? KS0_OFF : KS1_OFF;
      const bf16* src = Kh + ((size_t)bh * L_ + (c + 1) * 64 + sr) * D_ + ss;
      async_b128(dst + soff, src);
      async_b128(dst + soff + 16u, src + 8);
      wait_async2();   // oldest 2 (chunk c) are done; chunk c+1 in flight
    } else {
      wait_async0();
    }
    __syncthreads();
    const bf16* Kbuf = (c & 1) ? Ks1 : Ks0;
#pragma unroll
    for (int ct = 0; ct < 2; ++ct)
#pragma unroll
      for (int ks = 0; ks < 2; ++ks) {
        v16bf fa = load_frag(Qs + (wrt * 16) * 72 + ks * 32, 72, lane);
        v16bf fb = load_frag(Kbuf + (wch * 32 + ct * 16) * 72 + ks * 32, 72, lane);
        sc[c * 2 + ct] = wmma_bf16(fa, fb, sc[c * 2 + ct]);
      }
    __syncthreads();
  }

  // ---- logits = (scores + rel_skew)/8 + mask; row max --------------------
  const float scale = 0.125f;  // 1/sqrt(64)
  float rmax[8];
#pragma unroll
  for (int r = 0; r < 8; ++r) rmax[r] = -3.0e38f;
#pragma unroll
  for (int t = 0; t < 12; ++t) {
    int c = t >> 1, ct = t & 1;
    int j = c * 64 + wch * 32 + ct * 16 + nl;
    float pm = padm[j] * -1.0e9f;
#pragma unroll
    for (int r = 0; r < 8; ++r) {
      int il = wrt * 16 + hh + r;
      int ig = lt * 64 + il;
      int dm = j - ig;
      dm = dm < -31 ? -31 : (dm > 31 ? 31 : dm);
      float val = (sc[t][r] + rel_s[il * 64 + (dm + 31)]) * scale + pm;
      sc[t][r] = val;
      rmax[r] = fmaxf(rmax[r], val);
    }
  }
#pragma unroll
  for (int r = 0; r < 8; ++r)
#pragma unroll
    for (int off = 8; off > 0; off >>= 1)
      rmax[r] = fmaxf(rmax[r], __shfl_xor(rmax[r], off, 16));
  if (nl == 0)
#pragma unroll
    for (int r = 0; r < 8; ++r)
      red[(wrt * 16 + hh + r) * 2 + wch] = rmax[r];
  __syncthreads();
  float fmx[8];
#pragma unroll
  for (int r = 0; r < 8; ++r) {
    int il = wrt * 16 + hh + r;
    fmx[r] = fmaxf(red[il * 2 + 0], red[il * 2 + 1]);
  }
  __syncthreads();

  // ---- exp + row sum -----------------------------------------------------
  float rsum[8];
#pragma unroll
  for (int r = 0; r < 8; ++r) rsum[r] = 0.f;
#pragma unroll
  for (int t = 0; t < 12; ++t)
#pragma unroll
    for (int r = 0; r < 8; ++r) {
      float e = __expf(sc[t][r] - fmx[r]);
      sc[t][r] = e;
      rsum[r] += e;
    }
#pragma unroll
  for (int r = 0; r < 8; ++r)
#pragma unroll
    for (int off = 8; off > 0; off >>= 1)
      rsum[r] += __shfl_xor(rsum[r], off, 16);
  if (nl == 0)
#pragma unroll
    for (int r = 0; r < 8; ++r)
      red[(wrt * 16 + hh + r) * 2 + wch] = rsum[r];
  __syncthreads();
  float rinv[8];
#pragma unroll
  for (int r = 0; r < 8; ++r) {
    int il = wrt * 16 + hh + r;
    rinv[r] = 1.0f / (red[il * 2 + 0] + red[il * 2 + 1]);
  }

  // ---- attn out (fp32 to d_out, bf16 to LDS) -----------------------------
  float* abase = attn_out + ((size_t)bh * L_ + lt * 64) * L_;
#pragma unroll
  for (int t = 0; t < 12; ++t) {
    int c = t >> 1, ct = t & 1;
    int j = c * 64 + wch * 32 + ct * 16 + nl;
#pragma unroll
    for (int r = 0; r < 8; ++r) {
      int il = wrt * 16 + hh + r;
      float a = sc[t][r] * rinv[r];
      abase[(size_t)il * L_ + j] = a;
      attn_s[il * 392 + j] = (bf16)a;
    }
  }
  __syncthreads();

  // ---- relw[i][m]: bucketized attention mass (reuse rel_s) ---------------
  for (int idx = tid; idx < 64 * 64; idx += 256) rel_s[idx] = 0.f;
  __syncthreads();
  for (int idx = tid; idx < 64 * 61; idx += 256) {   // middle buckets 1..61
    int i = idx / 61, m = idx % 61 + 1;
    int j = lt * 64 + i + m - 31;
    rel_s[i * 64 + m] = (j >= 0 && j < L_) ? (float)attn_s[i * 392 + j] : 0.f;
  }
  for (int idx = tid; idx < 128; idx += 256) {        // edge buckets 0 and 62
    int i = idx >> 1, e = idx & 1;
    int ig = lt * 64 + i;
    float s = 0.f;
    if (e == 0) {
      int jmax = ig - 31;
      for (int j = 0; j <= jmax; ++j) s += (float)attn_s[i * 392 + j];
      rel_s[i * 64 + 0] = s;
    } else {
      for (int j = ig + 31; j < L_; ++j) s += (float)attn_s[i * 392 + j];
      rel_s[i * 64 + 62] = s;
    }
  }
  __syncthreads();

  // ---- out = attn @ V  (+ relw @ value_rel), V chunks double-buffered ----
  v8f oacc[2];
  oacc[0] = zero8();
  oacc[1] = zero8();
  const int octb = wch * 2;   // this wave's 2 output-column tiles

  {  // prologue: chunk 0 -> Vts0
    const bf16* src = Vt + ((size_t)bh * D_ + sr) * L_ + 0 * 64 + ss;
    async_b128(VTS0_OFF + soff, src);
    async_b128(VTS0_OFF + soff + 16u, src + 8);
  }
  for (int c = 0; c < 6; ++c) {
    if (c + 1 < 6) {
      unsigned dst = (c & 1) ? VTS0_OFF : VTS1_OFF;
      const bf16* src = Vt + ((size_t)bh * D_ + sr) * L_ + (c + 1) * 64 + ss;
      async_b128(dst + soff, src);
      async_b128(dst + soff + 16u, src + 8);
      wait_async2();
    } else {
      wait_async0();
    }
    __syncthreads();
    const bf16* Vbuf = (c & 1) ? Vts1 : Vts0;
#pragma unroll
    for (int ct = 0; ct < 2; ++ct)
#pragma unroll
      for (int ks = 0; ks < 2; ++ks) {
        v16bf fa = load_frag(attn_s + (wrt * 16) * 392 + c * 64 + ks * 32, 392, lane);
        v16bf fb = load_frag(Vbuf + ((octb + ct) * 16) * 72 + ks * 32, 72, lane);
        oacc[ct] = wmma_bf16(fa, fb, oacc[ct]);
      }
    __syncthreads();
  }
#pragma unroll
  for (int ct = 0; ct < 2; ++ct)
#pragma unroll
    for (int ks = 0; ks < 2; ++ks) {
      v16bf fa = load_frag_f32(rel_s + (wrt * 16) * 64 + ks * 32, 64, lane);
      v16bf fb = load_frag(VRTs + ((octb + ct) * 16) * 72 + ks * 32, 72, lane);
      oacc[ct] = wmma_bf16(fa, fb, oacc[ct]);
    }

#pragma unroll
  for (int ct = 0; ct < 2; ++ct)
#pragma unroll
    for (int r = 0; r < 8; ++r) {
      int il = wrt * 16 + hh + r;
      int Rg = b * L_ + lt * 64 + il;
      int Cg = h * 64 + (octb + ct) * 16 + nl;
      O[(size_t)Rg * DM_ + Cg] = (bf16)oacc[ct][r];
    }
}

// ---------------------------------------------------------------------------
// Kernel 3: output projection O(1536x1024,bf16) @ Wo + bo -> fp32 d_out.
// A-tile (bf16 raw bytes) staged async; Wo tile converted on the VALU path.
// Dynamic LDS: As @ 0 (128x40 bf16), Bt @ 10240 (128x40 bf16) = 20480 B.
// ---------------------------------------------------------------------------
#define AS3_OFF 0u
#define BT3_OFF 10240u
#define SMEM3_BYTES 20480u

__global__ __launch_bounds__(256) void outproj_kernel(
    const bf16* __restrict__ O, const float* __restrict__ Wo,
    const float* __restrict__ bo, float* __restrict__ out) {
  extern __shared__ unsigned char smem[];
  bf16* As = (bf16*)(smem + AS3_OFF);
  bf16* Bt = (bf16*)(smem + BT3_OFF);

  const int row0 = blockIdx.y * 128;
  const int col0 = blockIdx.x * 128;
  const int tid  = threadIdx.x;
  const int lane = tid & 31;
  const int w    = tid >> 5;
  const int wr   = w >> 2, wc = w & 3;

  v8f acc[4][2];
#pragma unroll
  for (int i = 0; i < 4; ++i)
#pragma unroll
    for (int j = 0; j < 2; ++j) acc[i][j] = zero8();

  for (int k0 = 0; k0 < DM_; k0 += 32) {
    {  // A tile 128x32 bf16: raw async copy (2 x b128 per thread)
      int r = tid >> 1, s = (tid & 1) * 16;
      const bf16* src = O + (size_t)(row0 + r) * DM_ + k0 + s;
      unsigned dst = AS3_OFF + (unsigned)(r * 40 + s) * 2;
      async_b128(dst, src);
      async_b128(dst + 16u, src + 8);
    }
    {  // B tile 32x128 fp32 -> bf16, transposed into Bt[n][k]
      int kk = tid >> 3, s = (tid & 7) * 16;
      const float* src = Wo + (size_t)(k0 + kk) * DM_ + col0 + s;
#pragma unroll
      for (int i = 0; i < 16; ++i) Bt[(s + i) * 40 + kk] = (bf16)src[i];
      if (k0 + 32 < DM_) __builtin_prefetch(src + 32 * DM_, 0, 0);
    }
    wait_async0();
    __syncthreads();
#pragma unroll
    for (int i = 0; i < 4; ++i) {
      v16bf fa = load_frag(As + (wr * 64 + i * 16) * 40, 40, lane);
#pragma unroll
      for (int j = 0; j < 2; ++j) {
        v16bf fb = load_frag(Bt + (wc * 32 + j * 16) * 40, 40, lane);
        acc[i][j] = wmma_bf16(fa, fb, acc[i][j]);
      }
    }
    __syncthreads();
  }

  const int hh = (lane & 16) >> 1;
  const int nl = lane & 15;
#pragma unroll
  for (int i = 0; i < 4; ++i)
#pragma unroll
    for (int j = 0; j < 2; ++j)
#pragma unroll
      for (int r = 0; r < 8; ++r) {
        int Rg = row0 + wr * 64 + i * 16 + r + hh;
        int Cg = col0 + wc * 32 + j * 16 + nl;
        out[(size_t)Rg * DM_ + Cg] = acc[i][j][r] + bo[Cg];
      }
}

// ---------------------------------------------------------------------------
extern "C" void kernel_launch(void* const* d_in, const int* in_sizes, int n_in,
                              void* d_out, int out_size, void* d_ws,
                              size_t ws_size, hipStream_t stream) {
  const float* v         = (const float*)d_in[0];
  const float* k         = (const float*)d_in[1];
  const float* q         = (const float*)d_in[2];
  const float* pad_mask  = (const float*)d_in[3];
  const float* Wq        = (const float*)d_in[4];
  const float* Wk        = (const float*)d_in[5];
  const float* Wv        = (const float*)d_in[6];
  const float* Wo        = (const float*)d_in[7];
  const float* bo        = (const float*)d_in[8];
  const float* key_rel   = (const float*)d_in[9];
  const float* value_rel = (const float*)d_in[10];

  float* out0     = (float*)d_out;                     // (B,L,DM)
  float* out_attn = out0 + (size_t)B_ * L_ * DM_;      // (B,H,L,L)

  const size_t nBH = (size_t)B_ * H_ * L_ * D_;        // per-buffer bf16 count
  bf16* Qh = (bf16*)d_ws;
  bf16* Kh = Qh + nBH;
  bf16* Vt = Kh + nBH;
  bf16* O  = Vt + nBH;                                 // (B*L, DM) bf16

  dim3 blk(256);
  proj_kernel<<<dim3(DM_ / 128, (B_ * L_) / 128, 3), blk, 0, stream>>>(
      q, k, v, Wq, Wk, Wv, Qh, Kh, Vt);

  attn_kernel<<<dim3(L_ / 64, H_, B_), blk, SMEM2_BYTES, stream>>>(
      Qh, Kh, Vt, key_rel, value_rel, pad_mask, O, out_attn);

  outproj_kernel<<<dim3(DM_ / 128, (B_ * L_) / 128), blk, SMEM3_BYTES, stream>>>(
      O, Wo, bo, out0);
}
